// StockMixer_90486370992825
// MI455X (gfx1250) — matrix operand order repaired and checked by reading, compiled, not run on hardware
//
#include <hip/hip_runtime.h>

typedef __attribute__((ext_vector_type(16))) _Float16 v16h;
typedef __attribute__((ext_vector_type(8)))  float    v8f;
typedef __attribute__((ext_vector_type(4)))  float    v4f;

#define NSTOCK 5000
#define MIXD   112

__device__ __forceinline__ float hswish(float x) {
  float t = fminf(fmaxf(x + 3.0f, 0.0f), 6.0f);
  return x * t * (1.0f / 6.0f);
}

// A-operand fragment from row-major f16 LDS buffer [rows][pitch].
// A 16x32 in the contiguous k order this code uses: lanes 0-15: M=lane, K=0..15;
// lanes 16-31: M=lane-16, K=16..31.
__device__ __forceinline__ v16h load_a_lds(const _Float16* base, int pitch,
                                           int mb, int kb, int lane) {
  int row = mb * 16 + (lane & 15);
  int hi  = (lane >> 4) & 1;
  const _Float16* p = base + row * pitch + kb * 32 + hi * 16;
  union { v16h v; v4f q[2]; } u;
  u.q[0] = *(const v4f*)(p);
  u.q[1] = *(const v4f*)(p + 8);
  return u.v;
}

// B-operand fragment from K-major (transposed) f16 LDS buffer [cols][64].
// B 32x16 in the contiguous k order this code uses (same as A): lanes 0-15: N=lane, K=0..15; lanes 16-31: N=lane-16, K=16..31.
__device__ __forceinline__ v16h load_b_ldsT(const _Float16* base,
                                            int nb, int kb, int lane) {
  int col = nb * 16 + (lane & 15);
  int hi  = (lane >> 4) & 1;
  const _Float16* p = base + col * 64 + kb * 32 + hi * 16;
  union { v16h v; v4f q[2]; } u;
  u.q[0] = *(const v4f*)(p);
  u.q[1] = *(const v4f*)(p + 8);
  return u.v;
}

__device__ __forceinline__ v8f wmma_f16(v16h a, v16h b, v8f c) {
  return __builtin_amdgcn_wmma_f32_16x16x32_f16(false, a, false, b,
                                                (short)0, c, false, false);
}

// ---------------- weight packing ----------------
// W [K][Nn] f32 -> fragment-packed f16: frag (kb,nb), lane L, half i:
//   col = nb*16 + (L&15), k = kb*32 + 16*(L>=16) + i
__global__ void pack_w(const float* __restrict__ W, _Float16* __restrict__ out,
                       int K, int Nn) {
  int e = blockIdx.x * blockDim.x + threadIdx.x;
  if (e >= K * Nn) return;
  int frag = e >> 9, w = e & 511;
  int lane = w >> 4, i = w & 15;
  int ntiles = Nn >> 4;
  int kb = frag / ntiles, nb = frag - kb * ntiles;
  int col = nb * 16 + (lane & 15);
  int kk  = kb * 32 + ((lane >= 16) ? 16 : 0) + i;
  out[e] = (_Float16)W[kk * Nn + col];
}

// U [Mdim][Kdim] f32 -> A-layout fragments, K zero-padded to multiples of 32.
__global__ void pack_u(const float* __restrict__ U, _Float16* __restrict__ out,
                       int Mdim, int Kdim) {
  int Ktp = (Kdim >= 32) ? (Kdim >> 5) : 1;
  int total = (Mdim >> 4) * Ktp * 512;
  int e = blockIdx.x * blockDim.x + threadIdx.x;
  if (e >= total) return;
  int frag = e >> 9, w = e & 511;
  int lane = w >> 4, i = w & 15;
  int mb = frag / Ktp, kb = frag - mb * Ktp;
  int row = mb * 16 + (lane & 15);
  int hi  = (lane >= 16) ? 16 : 0;
  int klocal = hi + i;
  int kk = kb * 32 + klocal;
  float v = (kk < Kdim) ? U[row * Kdim + kk] : 0.0f;
  out[e] = (_Float16)v;
}

// ---------------- main per-sample kernel ----------------
struct ScaleArgs {
  const float *g1, *b1, *g2, *b2, *fw, *fb;
  const _Float16 *w1p, *w2p, *u1p, *u2p;
};
struct MainArgs {
  const float* x;
  float* hcat; float* bsum; float* bsumsq;
  ScaleArgs sc[3];
};

__device__ __forceinline__ void block_stats(const float* buf, int n, float invn,
                                            float* sRed, float* sRed2,
                                            float* sStat, int tid) {
  float s = 0.f, s2 = 0.f;
  for (int i = tid; i < n; i += 256) { float v = buf[i]; s += v; s2 += v * v; }
  sRed[tid] = s; sRed2[tid] = s2;
  __syncthreads();
  for (int st = 128; st > 0; st >>= 1) {
    if (tid < st) { sRed[tid] += sRed[tid + st]; sRed2[tid] += sRed2[tid + st]; }
    __syncthreads();
  }
  if (tid == 0) {
    float m = sRed[0] * invn;
    float var = sRed2[0] * invn - m * m;
    sStat[0] = m;
    sStat[1] = rsqrtf(var + 1e-5f);
  }
  __syncthreads();
}

__global__ __launch_bounds__(256) void mixer_main(MainArgs A) {
  __shared__ float    sX[64 * 128];       // pooled input xp            (32 KB)
  __shared__ float    sW[64 * 128];       // xi / xt working buffer     (32 KB)
  __shared__ _Float16 sA16[64 * 128];     // f16 A operand (LN1 out)    (16 KB)
  __shared__ _Float16 sH16[64 * 256];     // MLP hidden; aliased h2T|uT (32 KB)
  __shared__ float    sRed[256], sRed2[256], sStat[2], sHc[128];

  const int n    = blockIdx.x;
  const int tid  = threadIdx.x;
  const int wave = tid >> 5;
  const int lane = tid & 31;
  const float* gx = A.x + (size_t)n * 64 * 128;

  int off = 0;
  for (int s = 0; s < 3; ++s) {
    const int kk = 1 << s;
    const int Tk = 64 >> s;
    const int TE = Tk * 128;
    const int Mt = Tk >> 4;
    const ScaleArgs& P = A.sc[s];

    // ---- avg-pool over time ----
    for (int idx = tid; idx < TE; idx += 256) {
      int t = idx >> 7, f = idx & 127;
      float acc = 0.f;
      for (int j = 0; j < kk; ++j) acc += gx[(t * kk + j) * 128 + f];
      sX[idx] = acc * (1.0f / kk);
    }
    __syncthreads();

    // ---- LayerNorm1 over whole [Tk,F] tile -> f16 A operand ----
    block_stats(sX, TE, 1.0f / TE, sRed, sRed2, sStat, tid);
    {
      float m = sStat[0], r = sStat[1];
      for (int idx = tid; idx < TE; idx += 256)
        sA16[idx] = (_Float16)((sX[idx] - m) * r * P.g1[idx] + P.b1[idx]);
    }
    __syncthreads();

    // ---- matmul1: hardswish(LN(xp) @ W1) -> sH16 [Tk,256] f16 ----
    for (int tile = wave; tile < Mt * 16; tile += 8) {
      int mb = tile >> 4, nb = tile & 15;
      v8f c = {};
      for (int kb = 0; kb < 4; ++kb) {
        v16h a = load_a_lds(sA16, 128, mb, kb, lane);
        v16h b = *(const v16h*)(P.w1p + (size_t)((kb * 16 + nb) * 32 + lane) * 16);
        c = wmma_f16(a, b, c);
      }
      int ncol  = nb * 16 + (lane & 15);
      int mbase = mb * 16 + ((lane >= 16) ? 8 : 0);
      for (int r = 0; r < 8; ++r)
        sH16[(mbase + r) * 256 + ncol] = (_Float16)hswish(c[r]);
    }
    __syncthreads();

    // ---- matmul2: xi = xp + (hidden @ W2) -> sW [Tk,128] f32 ----
    for (int tile = wave; tile < Mt * 8; tile += 8) {
      int mb = tile >> 3, nb = tile & 7;
      v8f c = {};
      for (int kb = 0; kb < 8; ++kb) {
        v16h a = load_a_lds(sH16, 256, mb, kb, lane);
        v16h b = *(const v16h*)(P.w2p + (size_t)((kb * 8 + nb) * 32 + lane) * 16);
        c = wmma_f16(a, b, c);
      }
      int ncol  = nb * 16 + (lane & 15);
      int mbase = mb * 16 + ((lane >= 16) ? 8 : 0);
      for (int r = 0; r < 8; ++r) {
        int ix = (mbase + r) * 128 + ncol;
        sW[ix] = sX[ix] + c[r];
      }
    }
    __syncthreads();

    // ---- LayerNorm2 -> transposed f16 buffer h2T [128][64] ----
    block_stats(sW, TE, 1.0f / TE, sRed, sRed2, sStat, tid);
    _Float16* h2T = sH16;             // [128][64] K-major
    _Float16* uT  = sH16 + 128 * 64;  // [128][64] K-major
    if (Tk < 32) {  // zero-pad K range [Tk,32) for the K=32 WMMA sweep
      for (int idx = tid; idx < 128 * 16; idx += 256) {
        int f = idx >> 4, t = 16 + (idx & 15);
        h2T[f * 64 + t] = (_Float16)0.f;
        uT[f * 64 + t]  = (_Float16)0.f;
      }
    }
    {
      float m = sStat[0], r = sStat[1];
      for (int idx = tid; idx < TE; idx += 256) {
        int t = idx >> 7, f = idx & 127;
        h2T[f * 64 + t] = (_Float16)((sW[idx] - m) * r * P.g2[idx] + P.b2[idx]);
      }
    }
    __syncthreads();

    // ---- matmul3: u = hardswish(U1 @ h2) -> uT (transposed) ----
    const int Ktp = (Tk == 64) ? 2 : 1;
    for (int tile = wave; tile < Mt * 8; tile += 8) {
      int mb = tile >> 3, nb = tile & 7;
      v8f c = {};
      for (int kb = 0; kb < Ktp; ++kb) {
        v16h a = *(const v16h*)(P.u1p + (size_t)((mb * Ktp + kb) * 32 + lane) * 16);
        v16h b = load_b_ldsT(h2T, nb, kb, lane);
        c = wmma_f16(a, b, c);
      }
      int f     = nb * 16 + (lane & 15);
      int tbase = mb * 16 + ((lane >= 16) ? 8 : 0);
      for (int r = 0; r < 8; ++r)
        uT[f * 64 + tbase + r] = (_Float16)hswish(c[r]);
    }
    __syncthreads();

    // ---- matmul4: xt = xi + (U2 @ u) -> sW in place ----
    for (int tile = wave; tile < Mt * 8; tile += 8) {
      int mb = tile >> 3, nb = tile & 7;
      v8f c = {};
      for (int kb = 0; kb < Ktp; ++kb) {
        v16h a = *(const v16h*)(P.u2p + (size_t)((mb * Ktp + kb) * 32 + lane) * 16);
        v16h b = load_b_ldsT(uT, nb, kb, lane);
        c = wmma_f16(a, b, c);
      }
      int f     = nb * 16 + (lane & 15);
      int tbase = mb * 16 + ((lane >= 16) ? 8 : 0);
      for (int r = 0; r < 8; ++r) {
        int ix = (tbase + r) * 128 + f;
        sW[ix] = sW[ix] + c[r];
      }
    }
    __syncthreads();

    // ---- fc(F -> 1): per-time-step dot with fw ----
    {
      float fb = P.fb[0];
      for (int t = tid; t < Tk; t += 256) {
        float acc = 0.f;
        for (int f = 0; f < 128; ++f) acc += sW[t * 128 + f] * P.fw[f];
        sHc[off + t] = acc + fb;
      }
    }
    __syncthreads();
    off += Tk;
  }

  // ---- emit hcat row + partial sums for global LayerNorm ----
  if (tid < MIXD) A.hcat[(size_t)n * MIXD + tid] = sHc[tid];
  float v = (tid < MIXD) ? sHc[tid] : 0.f;
  sRed[tid] = v; sRed2[tid] = v * v;
  __syncthreads();
  for (int st = 128; st > 0; st >>= 1) {
    if (tid < st) { sRed[tid] += sRed[tid + st]; sRed2[tid] += sRed2[tid + st]; }
    __syncthreads();
  }
  if (tid == 0) { A.bsum[n] = sRed[0]; A.bsumsq[n] = sRed2[0]; }
}

// ---------------- cross-stock mix ----------------
__global__ void reduce_stats(const float* __restrict__ bsum,
                             const float* __restrict__ bsumsq,
                             float* __restrict__ stats) {
  __shared__ float r1[256], r2[256];
  int tid = threadIdx.x;
  float s = 0.f, s2 = 0.f;
  for (int i = tid; i < NSTOCK; i += 256) { s += bsum[i]; s2 += bsumsq[i]; }
  r1[tid] = s; r2[tid] = s2;
  __syncthreads();
  for (int st = 128; st > 0; st >>= 1) {
    if (tid < st) { r1[tid] += r1[tid + st]; r2[tid] += r2[tid + st]; }
    __syncthreads();
  }
  if (tid == 0) {
    float inv = 1.0f / ((float)NSTOCK * (float)MIXD);
    float m = r1[0] * inv;
    float var = r2[0] * inv - m * m;
    stats[0] = m; stats[1] = rsqrtf(var + 1e-5f);
  }
}

__global__ void mix1(const float* __restrict__ hcat, const float* __restrict__ gm,
                     const float* __restrict__ bm, const float* __restrict__ M1,
                     const float* __restrict__ stats, float* __restrict__ G) {
  int m = blockIdx.x, j = threadIdx.x;
  if (j >= MIXD) return;
  float mu = stats[0], rs = stats[1];
  float acc = 0.f;
  const float* M1r = M1 + (size_t)m * NSTOCK;
  for (int n2 = 0; n2 < NSTOCK; ++n2) {
    size_t ix = (size_t)n2 * MIXD + j;
    float hm = (hcat[ix] - mu) * rs * gm[ix] + bm[ix];
    acc += M1r[n2] * hm;
  }
  G[m * MIXD + j] = hswish(acc);
}

__global__ void finalk(const float* __restrict__ hcat, const float* __restrict__ M2,
                       const float* __restrict__ G, const float* __restrict__ fw2,
                       const float* __restrict__ fb2, float* __restrict__ out) {
  __shared__ float red[128];
  int n = blockIdx.x, j = threadIdx.x;
  float p = 0.f;
  if (j < MIXD) {
    float hm = 0.f;
    for (int m = 0; m < 32; ++m) hm += M2[(size_t)n * 32 + m] * G[m * MIXD + j];
    float h = hcat[(size_t)n * MIXD + j];
    p = h * fw2[j] + (h + hm) * fw2[MIXD + j];
  }
  red[j] = p;
  __syncthreads();
  for (int st = 64; st > 0; st >>= 1) {
    if (j < st) red[j] += red[j + st];
    __syncthreads();
  }
  if (j == 0) out[n] = red[0] + fb2[0];
}

// ---------------- launch ----------------
extern "C" void kernel_launch(void* const* d_in, const int* in_sizes, int n_in,
                              void* d_out, int out_size, void* d_ws, size_t ws_size,
                              hipStream_t stream) {
  (void)in_sizes; (void)n_in; (void)out_size; (void)ws_size;
  char* ws = (char*)d_ws;
  const size_t OFF_W1 = 0;                      // 3 * 65536
  const size_t OFF_W2 = 3 * 65536;              // 3 * 65536
  const size_t OFF_U  = 6 * 65536;              // 6 * 16384
  const size_t OFF_HC = OFF_U + 6 * 16384;      // 5000*112*4
  const size_t OFF_BS = OFF_HC + (size_t)NSTOCK * MIXD * 4;
  const size_t OFF_BQ = OFF_BS + 20480;
  const size_t OFF_ST = OFF_BQ + 20480;
  const size_t OFF_G  = OFF_ST + 256;

  MainArgs ma;
  ma.x      = (const float*)d_in[0];
  ma.hcat   = (float*)(ws + OFF_HC);
  ma.bsum   = (float*)(ws + OFF_BS);
  ma.bsumsq = (float*)(ws + OFF_BQ);

  for (int s = 0; s < 3; ++s) {
    int base = 1 + s * 10;
    const float* W1 = (const float*)d_in[base + 2];
    const float* W2 = (const float*)d_in[base + 3];
    const float* U1 = (const float*)d_in[base + 6];
    const float* U2 = (const float*)d_in[base + 7];
    int Tk = 64 >> s;
    _Float16* w1p = (_Float16*)(ws + OFF_W1 + (size_t)s * 65536);
    _Float16* w2p = (_Float16*)(ws + OFF_W2 + (size_t)s * 65536);
    _Float16* u1p = (_Float16*)(ws + OFF_U + (size_t)s * 16384);
    _Float16* u2p = (_Float16*)(ws + OFF_U + (size_t)(3 + s) * 16384);

    pack_w<<<(128 * 256 + 255) / 256, 256, 0, stream>>>(W1, w1p, 128, 256);
    pack_w<<<(256 * 128 + 255) / 256, 256, 0, stream>>>(W2, w2p, 256, 128);
    int Ktp  = (Tk == 64) ? 2 : 1;
    int totU = (Tk / 16) * Ktp * 512;
    pack_u<<<(totU + 255) / 256, 256, 0, stream>>>(U1, u1p, Tk, Tk);
    pack_u<<<(totU + 255) / 256, 256, 0, stream>>>(U2, u2p, Tk, Tk);

    ma.sc[s].g1 = (const float*)d_in[base + 0];
    ma.sc[s].b1 = (const float*)d_in[base + 1];
    ma.sc[s].g2 = (const float*)d_in[base + 4];
    ma.sc[s].b2 = (const float*)d_in[base + 5];
    ma.sc[s].fw = (const float*)d_in[base + 8];
    ma.sc[s].fb = (const float*)d_in[base + 9];
    ma.sc[s].w1p = w1p; ma.sc[s].w2p = w2p;
    ma.sc[s].u1p = u1p; ma.sc[s].u2p = u2p;
  }

  mixer_main<<<NSTOCK, 256, 0, stream>>>(ma);
  reduce_stats<<<1, 256, 0, stream>>>((const float*)(ws + OFF_BS),
                                      (const float*)(ws + OFF_BQ),
                                      (float*)(ws + OFF_ST));
  mix1<<<32, 128, 0, stream>>>(ma.hcat, (const float*)d_in[31],
                               (const float*)d_in[32], (const float*)d_in[33],
                               (const float*)(ws + OFF_ST), (float*)(ws + OFF_G));
  finalk<<<NSTOCK, 128, 0, stream>>>(ma.hcat, (const float*)d_in[34],
                                     (const float*)(ws + OFF_G),
                                     (const float*)d_in[35],
                                     (const float*)d_in[36], (float*)d_out);
}
